// MambaBlock_74371653697794
// MI455X (gfx1250) — matrix operand order repaired and checked by reading, not compiled, not run on hardware
//
#include <hip/hip_runtime.h>
#include <hip/hip_bf16.h>
#include <cstdint>

#define D_MODEL   1024
#define D_INNER   2048
#define D_STATE   16
#define D_CONV    4
#define DT_RANK   64
#define NBATCH    2
#define SEQLEN    1024
#define NROWS     (NBATCH * SEQLEN)        // 2048 token rows
#define XZ_COLS   (2 * D_INNER)            // 4096
#define XDBL_COLS (DT_RANK + 2 * D_STATE)  // 96

typedef __attribute__((ext_vector_type(8)))  float  v8f;
typedef __attribute__((ext_vector_type(8)))  __bf16 v8bf;
typedef __attribute__((ext_vector_type(16))) __bf16 v16bf;

__device__ __forceinline__ v16bf cat8(v8bf a, v8bf b) {
  return __builtin_shufflevector(a, b, 0,1,2,3,4,5,6,7,8,9,10,11,12,13,14,15);
}

// A fragment (16x32 bf16, §7.12.2): lane reads K chunks [k0+8h .. +7] and [k0+16+8h .. +7]
// p already points at  row*lda + 8*half ; chunks are p+k0 and p+k0+16.
__device__ __forceinline__ v16bf load_afrag(const __bf16* p) {
  v8bf c0 = *(const v8bf*)(p);
  v8bf c1 = *(const v8bf*)(p + 16);
  return cat8(c0, c1);
}

__device__ __forceinline__ void split_f32(float v, __bf16& hi, __bf16& lo) {
  hi = (__bf16)v;
  lo = (__bf16)(v - (float)hi);
}

// ---------------------------------------------------------------------------
// Error-compensated bf16x3 WMMA GEMM:
//   C[M,N] = (Ahi+Alo)[M,K] * (Whi+Wlo)[N,K]^T  ~  Ahi*Whi + Ahi*Wlo + Alo*Whi
// f32 accumulation inside V_WMMA_F32_16X16X32_BF16. One wave -> 16x32 strip
// (2 accumulators, A fragment reused). K-loop is software-pipelined: next
// iteration's fragments are in flight while the 6 WMMAs of this one issue.
// Optional epilogue: f32 store and/or bf16 hi/lo split store (+bias[N]).
// ---------------------------------------------------------------------------
__global__ __launch_bounds__(256) void gemm_bf16x3(
    const __bf16* __restrict__ Ahi, const __bf16* __restrict__ Alo, int lda,
    const __bf16* __restrict__ Whi, const __bf16* __restrict__ Wlo, int ldw,
    const float* __restrict__ bias,
    float* __restrict__ Cf,
    __bf16* __restrict__ Chi, __bf16* __restrict__ Clo, int ldc,
    int M, int N, int K)
{
  const int wave   = (blockIdx.x * blockDim.x + threadIdx.x) >> 5; // wave-uniform
  const int lane   = threadIdx.x & 31;
  const int strips = N >> 5;                 // N / 32  (NT = 2)
  const int mTile  = wave / strips;
  const int nBase  = (wave % strips) * 32;
  if (mTile * 16 >= M) return;               // uniform -> EXEC stays all-ones

  const int half = lane >> 4;
  const int mr   = lane & 15;

  const __bf16* aH  = Ahi + (size_t)(mTile * 16 + mr) * lda + half * 8;
  const __bf16* aL  = Alo + (size_t)(mTile * 16 + mr) * lda + half * 8;
  const __bf16* b0H = Whi + (size_t)(nBase + mr)      * ldw + half * 8;
  const __bf16* b0L = Wlo + (size_t)(nBase + mr)      * ldw + half * 8;
  const __bf16* b1H = Whi + (size_t)(nBase + 16 + mr) * ldw + half * 8;
  const __bf16* b1L = Wlo + (size_t)(nBase + 16 + mr) * ldw + half * 8;

  v8f acc0 = {}, acc1 = {};

  v16bf ah  = load_afrag(aH), al = load_afrag(aL);
  v16bf bh0 = load_afrag(b0H), bl0 = load_afrag(b0L);
  v16bf bh1 = load_afrag(b1H), bl1 = load_afrag(b1L);

  for (int k0 = 32; k0 < K; k0 += 32) {
    // prefetch next fragments (issued before this step's WMMAs consume `cur`)
    v16bf ahn  = load_afrag(aH + k0);
    v16bf aln  = load_afrag(aL + k0);
    v16bf bh0n = load_afrag(b0H + k0);
    v16bf bl0n = load_afrag(b0L + k0);
    v16bf bh1n = load_afrag(b1H + k0);
    v16bf bl1n = load_afrag(b1L + k0);

    acc0 = __builtin_amdgcn_wmma_f32_16x16x32_bf16(false, ah, false, bh0, (short)0, acc0, false, false);
    acc1 = __builtin_amdgcn_wmma_f32_16x16x32_bf16(false, ah, false, bh1, (short)0, acc1, false, false);
    acc0 = __builtin_amdgcn_wmma_f32_16x16x32_bf16(false, ah, false, bl0, (short)0, acc0, false, false);
    acc1 = __builtin_amdgcn_wmma_f32_16x16x32_bf16(false, ah, false, bl1, (short)0, acc1, false, false);
    acc0 = __builtin_amdgcn_wmma_f32_16x16x32_bf16(false, al, false, bh0, (short)0, acc0, false, false);
    acc1 = __builtin_amdgcn_wmma_f32_16x16x32_bf16(false, al, false, bh1, (short)0, acc1, false, false);

    ah = ahn; al = aln; bh0 = bh0n; bl0 = bl0n; bh1 = bh1n; bl1 = bl1n;
  }
  // drain last stage
  acc0 = __builtin_amdgcn_wmma_f32_16x16x32_bf16(false, ah, false, bh0, (short)0, acc0, false, false);
  acc1 = __builtin_amdgcn_wmma_f32_16x16x32_bf16(false, ah, false, bh1, (short)0, acc1, false, false);
  acc0 = __builtin_amdgcn_wmma_f32_16x16x32_bf16(false, ah, false, bl0, (short)0, acc0, false, false);
  acc1 = __builtin_amdgcn_wmma_f32_16x16x32_bf16(false, ah, false, bl1, (short)0, acc1, false, false);
  acc0 = __builtin_amdgcn_wmma_f32_16x16x32_bf16(false, al, false, bh0, (short)0, acc0, false, false);
  acc1 = __builtin_amdgcn_wmma_f32_16x16x32_bf16(false, al, false, bh1, (short)0, acc1, false, false);

  // epilogue: C/D layout -> VGPR r : M = r + 8*half, N = lane%16
#pragma unroll
  for (int j = 0; j < 2; ++j) {
    const int col = nBase + j * 16 + mr;
    const float bv = bias ? bias[col] : 0.0f;
#pragma unroll
    for (int r = 0; r < 8; ++r) {
      const int row = mTile * 16 + r + half * 8;
      const float v = (j == 0 ? acc0[r] : acc1[r]) + bv;
      const size_t idx = (size_t)row * ldc + col;
      if (Cf)  Cf[idx] = v;
      if (Chi) { __bf16 h, lo; split_f32(v, h, lo); Chi[idx] = h; Clo[idx] = lo; }
    }
  }
}

// ---------------------------------------------------------------------------
// f32 -> (bf16 hi, bf16 lo) split, elementwise (weights / x).
// ---------------------------------------------------------------------------
__global__ __launch_bounds__(256) void split_kernel(
    const float* __restrict__ src, __bf16* __restrict__ hi,
    __bf16* __restrict__ lo, int n)
{
  const int i = blockIdx.x * blockDim.x + threadIdx.x;
  if (i < n) {
    __bf16 h, l; split_f32(src[i], h, l);
    hi[i] = h; lo[i] = l;
  }
}

// ---------------------------------------------------------------------------
// Depthwise causal conv (k=4, zero left-pad) + bias + SiLU.
// Emits u in f32 (for the scan) and bf16 hi/lo (for the x_dbl GEMM).
// ---------------------------------------------------------------------------
#define LCHUNK 256
__global__ __launch_bounds__(256) void conv_silu_kernel(
    const float* __restrict__ xz,
    const float* __restrict__ conv_w,
    const float* __restrict__ conv_b,
    float* __restrict__ uf, __bf16* __restrict__ uhi, __bf16* __restrict__ ulo)
{
  const int tid   = blockIdx.x * blockDim.x + threadIdx.x;
  const int d     = tid % D_INNER;
  const int rest  = tid / D_INNER;
  const int nch   = SEQLEN / LCHUNK;
  const int chunk = rest % nch;
  const int b     = rest / nch;
  if (b >= NBATCH) return;

  const float w0 = conv_w[d * D_CONV + 0];
  const float w1 = conv_w[d * D_CONV + 1];
  const float w2 = conv_w[d * D_CONV + 2];
  const float w3 = conv_w[d * D_CONV + 3];
  const float bs = conv_b[d];

  const int l0 = chunk * LCHUNK;
  const size_t rowBase = (size_t)b * SEQLEN;

  float xm3 = (l0 - 3 >= 0) ? xz[(rowBase + l0 - 3) * XZ_COLS + d] : 0.0f;
  float xm2 = (l0 - 2 >= 0) ? xz[(rowBase + l0 - 2) * XZ_COLS + d] : 0.0f;
  float xm1 = (l0 - 1 >= 0) ? xz[(rowBase + l0 - 1) * XZ_COLS + d] : 0.0f;

  for (int i = 0; i < LCHUNK; ++i) {
    const int l = l0 + i;
    const float cur = xz[(rowBase + l) * XZ_COLS + d];
    const float v = w0 * xm3 + w1 * xm2 + w2 * xm1 + w3 * cur + bs;
    const float s = v / (1.0f + __expf(-v));              // SiLU
    const size_t idx = (rowBase + l) * D_INNER + d;
    uf[idx] = s;
    __bf16 h, lo; split_f32(s, h, lo);
    uhi[idx] = h; ulo[idx] = lo;
    xm3 = xm2; xm2 = xm1; xm1 = cur;
  }
}

// ---------------------------------------------------------------------------
// Fused selective scan, state-parallel version: one lane per (b, d, s).
// 65536 lanes = 2048 waves (16x the parallelism of channel-per-lane).
// Each lane carries one scalar h; per timestep: 1 exp + 1 FMA, then the
// y = sum_s h*C reduction is a 4-step shfl_xor butterfly within each
// 16-lane half of the wave32 (xor masks 8/4/2/1 never cross the half).
// dt/u/z loads are uniform per 16-lane group (broadcast); B/C loads are
// 16 consecutive floats (coalesced). Lane s==0 applies D-skip + SiLU(z)
// gate and emits bf16 hi/lo for the output GEMM.
// ---------------------------------------------------------------------------
__global__ __launch_bounds__(256) void scan_kernel(
    const float* __restrict__ u,
    const float* __restrict__ dt,
    const float* __restrict__ xdbl,
    const float* __restrict__ xz,       // z = xz[:, D_INNER:]
    const float* __restrict__ A_log,
    const float* __restrict__ D_param,
    __bf16* __restrict__ yhi, __bf16* __restrict__ ylo)
{
  const int tid = blockIdx.x * blockDim.x + threadIdx.x;
  const int s   = tid & (D_STATE - 1);
  const int ch  = tid >> 4;            // global channel index = b*D_INNER + d
  const int d   = ch & (D_INNER - 1);
  const int b   = ch >> 11;            // / D_INNER
  if (b >= NBATCH) return;

  const float Aneg = -__expf(A_log[(size_t)d * D_STATE + s]);
  const float Dp   = D_param[d];

  float h = 0.0f;
  for (int l = 0; l < SEQLEN; ++l) {
    const size_t row = (size_t)b * SEQLEN + l;
    const float dtv = dt[row * D_INNER + d];            // group-uniform
    const float uv  = u [row * D_INNER + d];            // group-uniform
    const float Bv  = xdbl[row * XDBL_COLS + DT_RANK + s];
    const float Cv  = xdbl[row * XDBL_COLS + DT_RANK + D_STATE + s];

    const float dA = __expf(dtv * Aneg);                // no dep on h: pipelines
    h = dA * h + dtv * Bv * uv;                         // the only serial chain
    float p = h * Cv;
    p += __shfl_xor(p, 8, 32);
    p += __shfl_xor(p, 4, 32);
    p += __shfl_xor(p, 2, 32);
    p += __shfl_xor(p, 1, 32);
    if (s == 0) {
      const float zv = xz[row * XZ_COLS + D_INNER + d];
      const float y  = p + uv * Dp;
      const float sz = zv / (1.0f + __expf(-zv));       // SiLU(z)
      const float yv = y * sz;
      __bf16 hh, ll; split_f32(yv, hh, ll);
      yhi[row * D_INNER + d] = hh;
      ylo[row * D_INNER + d] = ll;
    }
  }
}

// ---------------------------------------------------------------------------
extern "C" void kernel_launch(void* const* d_in, const int* in_sizes, int n_in,
                              void* d_out, int out_size, void* d_ws, size_t ws_size,
                              hipStream_t stream) {
  (void)in_sizes; (void)n_in; (void)out_size; (void)ws_size;
  const float* x      = (const float*)d_in[0];
  const float* W_in   = (const float*)d_in[1];
  const float* conv_w = (const float*)d_in[2];
  const float* conv_b = (const float*)d_in[3];
  const float* W_x    = (const float*)d_in[4];
  const float* W_dt   = (const float*)d_in[5];
  const float* b_dt   = (const float*)d_in[6];
  const float* A_log  = (const float*)d_in[7];
  const float* D_par  = (const float*)d_in[8];
  const float* W_out  = (const float*)d_in[9];
  float* out = (float*)d_out;

  char* ws = (char*)d_ws;
  float*  xz      = (float*) (ws + 0);          // 2048*4096*4 = 33554432
  float*  uf      = (float*) (ws + 33554432);   // 2048*2048*4 = 16777216
  float*  xdbl    = (float*) (ws + 50331648);   // 2048*  96*4 =   786432
  float*  dt      = (float*) (ws + 51118080);   // 2048*2048*4 = 16777216
  __bf16* x_hi    = (__bf16*)(ws + 67895296);   // 2048*1024*2 =  4194304
  __bf16* x_lo    = (__bf16*)(ws + 72089600);
  __bf16* u_hi    = (__bf16*)(ws + 76283904);   // 2048*2048*2 =  8388608
  __bf16* u_lo    = (__bf16*)(ws + 84672512);
  __bf16* xd_hi   = (__bf16*)(ws + 93061120);   // 2048*  96*2 =   393216
  __bf16* xd_lo   = (__bf16*)(ws + 93454336);
  __bf16* y_hi    = (__bf16*)(ws + 93847552);   // 2048*2048*2 =  8388608
  __bf16* y_lo    = (__bf16*)(ws + 102236160);
  __bf16* Wi_hi   = (__bf16*)(ws + 110624768);  // 4096*1024*2 =  8388608
  __bf16* Wi_lo   = (__bf16*)(ws + 119013376);
  __bf16* Wx_hi   = (__bf16*)(ws + 127401984);  //   96*2048*2 =   393216
  __bf16* Wx_lo   = (__bf16*)(ws + 127795200);
  __bf16* Wdt_hi  = (__bf16*)(ws + 128188416);  // 2048*  64*2 =   262144
  __bf16* Wdt_lo  = (__bf16*)(ws + 128450560);
  __bf16* Wo_hi   = (__bf16*)(ws + 128712704);  // 1024*2048*2 =  4194304
  __bf16* Wo_lo   = (__bf16*)(ws + 132907008);

  // 0) split operands into bf16 hi/lo pairs
  split_kernel<<<8192, 256, 0, stream>>>(x,     x_hi,   x_lo,   NROWS * D_MODEL);
  split_kernel<<<16384,256, 0, stream>>>(W_in,  Wi_hi,  Wi_lo,  XZ_COLS * D_MODEL);
  split_kernel<<<768,  256, 0, stream>>>(W_x,   Wx_hi,  Wx_lo,  XDBL_COLS * D_INNER);
  split_kernel<<<512,  256, 0, stream>>>(W_dt,  Wdt_hi, Wdt_lo, D_INNER * DT_RANK);
  split_kernel<<<8192, 256, 0, stream>>>(W_out, Wo_hi,  Wo_lo,  D_MODEL * D_INNER);

  // 1) xz = x @ W_in^T       M=2048 N=4096 K=1024 -> 128*128 waves -> 2048 blocks
  gemm_bf16x3<<<2048, 256, 0, stream>>>(x_hi, x_lo, D_MODEL, Wi_hi, Wi_lo, D_MODEL,
                                        nullptr, xz, nullptr, nullptr, XZ_COLS,
                                        NROWS, XZ_COLS, D_MODEL);
  // 2) depthwise conv + SiLU -> u (f32 + bf16 pair)
  conv_silu_kernel<<<64, 256, 0, stream>>>(xz, conv_w, conv_b, uf, u_hi, u_lo);
  // 3) x_dbl = u @ W_x^T     M=2048 N=96 K=2048 -> 128*3 waves -> 48 blocks
  gemm_bf16x3<<<48, 256, 0, stream>>>(u_hi, u_lo, D_INNER, Wx_hi, Wx_lo, D_INNER,
                                      nullptr, xdbl, xd_hi, xd_lo, XDBL_COLS,
                                      NROWS, XDBL_COLS, D_INNER);
  // 4) dt = x_dbl[:, :64] @ W_dt^T + b_dt   M=2048 N=2048 K=64 lda=96 -> 1024 blocks
  gemm_bf16x3<<<1024, 256, 0, stream>>>(xd_hi, xd_lo, XDBL_COLS, Wdt_hi, Wdt_lo, DT_RANK,
                                        b_dt, dt, nullptr, nullptr, D_INNER,
                                        NROWS, D_INNER, DT_RANK);
  // 5) fused selective scan + gate, state-parallel: 65536 lanes -> 256 blocks
  scan_kernel<<<256, 256, 0, stream>>>(uf, dt, xdbl, xz, A_log, D_par, y_hi, y_lo);
  // 6) out = ygate @ W_out^T  M=2048 N=1024 K=2048 -> 128*32 waves -> 512 blocks
  gemm_bf16x3<<<512, 256, 0, stream>>>(y_hi, y_lo, D_INNER, Wo_hi, Wo_lo, D_INNER,
                                       nullptr, out, nullptr, nullptr, D_MODEL,
                                       NROWS, D_MODEL, D_INNER);
}
